// Word2Vec_61186104099428
// MI455X (gfx1250) — matrix operand order repaired and checked by reading, not compiled, not run on hardware
//
#include <hip/hip_runtime.h>
#include <hip/hip_bf16.h>
#include <math.h>

// ---------------- problem constants ----------------
#define VOCAB 100000
#define EMBED 128
#define BATCH 4096
#define CTX   20

#define NTILE_N        (VOCAB / 16)          // 6250 exact
#define TILES_PER_BLK  128                   // 8 waves x 16 iterations
#define NCHUNK         ((NTILE_N + TILES_PER_BLK - 1) / TILES_PER_BLK)  // 49
#define NPART          (NCHUNK * 8)          // partials per row = 392

#define LOG2E 1.44269504088896340736f

// raw hardware exp2: a single v_exp_f32, no denormal guard (softmax args are <= 0,
// deep-negative inputs underflow to 0 in hardware, which is exactly what we want)
#define EXP2(x) __builtin_amdgcn_exp2f(x)

typedef __attribute__((ext_vector_type(16))) __bf16 v16bf;
typedef __attribute__((ext_vector_type(8)))  __bf16 v8bf;
typedef __attribute__((ext_vector_type(8)))  float  v8f;

// stride (in bf16 elements) between consecutive n-tiles handled by one wave (8 tiles apart)
#define WAVE_TILE_STRIDE ((size_t)8 * 16 * EMBED)

// ---------------- phase 1: proj = mean(gather(in_table, ids)) * log2(e), bf16 --------
// Scaling A by log2(e) puts the whole softmax in base-2: every exponential in the
// downstream kernels is a bare v_exp_f32 with no log2e multiply.
__global__ void proj_kernel(const int* __restrict__ ids,
                            const float* __restrict__ in_table,
                            __bf16* __restrict__ projB) {
    int row = blockIdx.x;        // 0..4095
    int d   = threadIdx.x;       // 0..127
    const int* c = ids + row * CTX;
    float s = 0.f;
    #pragma unroll 4
    for (int j = 0; j < CTX; ++j)
        s += in_table[(size_t)c[j] * EMBED + d];
    projB[(size_t)row * EMBED + d] = (__bf16)(s * (LOG2E / CTX));
}

// ---------------- phase 1b: out_table f32 [K][N] -> bf16 [N][K] (LDS transpose) ------
__global__ void cvt_kernel(const float* __restrict__ out_table,
                           __bf16* __restrict__ outT) {
    __shared__ float tile[32][EMBED + 1];
    int n0 = blockIdx.x * 32;            // 3125 blocks
    int t  = threadIdx.x;                // 256 threads
    int tn = t & 31, tk = t >> 5;        // tk in 0..7
    #pragma unroll
    for (int kk = 0; kk < EMBED; kk += 8)
        tile[tn][kk + tk] = out_table[(size_t)(kk + tk) * VOCAB + n0 + tn];
    __syncthreads();
    int nl = t >> 3;                     // 0..31
    int kb = (t & 7) * 16;               // 0..112
    __bf16* dst = outT + (size_t)(n0 + nl) * EMBED + kb;
    #pragma unroll
    for (int j = 0; j < 16; ++j)
        dst[j] = (__bf16)tile[nl][kb + j];
}

// ---------------- WMMA fragment helpers (row/col per ISA, k contiguous per lane half, wave32) -----------------
// A 16x32 bf16: lane m (rows 0..15): VGPR0-7 K = base..base+15 (contiguous, matching B)
// where base = 0 for lanes 0..15, 16 for lanes 16..31.
__device__ __forceinline__ void load_afrag(const __bf16* __restrict__ projB,
                                           int m0, int lane, v16bf a[4]) {
    const bool hi = lane >= 16;
    const __bf16* aRow = projB + (size_t)(m0 + (lane & 15)) * EMBED;
    const int off = hi ? 16 : 0;
    #pragma unroll
    for (int t = 0; t < 4; ++t) {
        v8bf lo = *(const v8bf*)(aRow + 32 * t + off);
        v8bf hh = *(const v8bf*)(aRow + 32 * t + 8 + off);
        v16bf f;
        #pragma unroll
        for (int j = 0; j < 8; ++j) { f[j] = lo[j]; f[8 + j] = hh[j]; }
        a[t] = f;
    }
}

// B 32x16 bf16: lane n holds column n, K = 0..15 (lanes 0..15) or 16..31 (lanes 16..31),
// contiguous in the pre-transposed [N][K] bf16 table. 128 bytes per lane per tile.
// Load all 4 fragments as one clause, then issue the 4 WMMAs (staged waits).
__device__ __forceinline__ v8f tile_gemm(const v16bf a[4],
                                         const __bf16* __restrict__ colPtr,
                                         int koff) {
    v16bf b0 = *(const v16bf*)(colPtr +  0 + koff);
    v16bf b1 = *(const v16bf*)(colPtr + 32 + koff);
    v16bf b2 = *(const v16bf*)(colPtr + 64 + koff);
    v16bf b3 = *(const v16bf*)(colPtr + 96 + koff);
    v8f c = {};
    c = __builtin_amdgcn_wmma_f32_16x16x32_bf16(false, a[0], false, b0, (short)0, c, false, false);
    c = __builtin_amdgcn_wmma_f32_16x16x32_bf16(false, a[1], false, b1, (short)0, c, false, false);
    c = __builtin_amdgcn_wmma_f32_16x16x32_bf16(false, a[2], false, b2, (short)0, c, false, false);
    c = __builtin_amdgcn_wmma_f32_16x16x32_bf16(false, a[3], false, b3, (short)0, c, false, false);
    return c;
}

// ---------------- phase 2: per (row, chunk, wave) online softmax partials ------------
__global__ void partial_kernel(const __bf16* __restrict__ projB,
                               const __bf16* __restrict__ outT,
                               float2* __restrict__ partials) {
    int lane = threadIdx.x & 31;
    int wave = threadIdx.x >> 5;          // 0..7
    int m0   = blockIdx.y * 16;           // row tile
    const bool hi  = lane >= 16;
    const int koff = hi ? 16 : 0;
    const int col0 = lane & 15;

    v16bf a[4];
    load_afrag(projB, m0, lane, a);

    float mrun[8], srun[8];
    #pragma unroll
    for (int r = 0; r < 8; ++r) { mrun[r] = -INFINITY; srun[r] = 0.f; }

    int base = blockIdx.x * TILES_PER_BLK + wave;
    int nt   = (base < NTILE_N) ? min(16, (NTILE_N - base + 7) / 8) : 0;

    const __bf16* colPtr = outT + (size_t)(base * 16 + col0) * EMBED;

    #pragma unroll 1
    for (int i = 0; i < nt; ++i) {
        if (i + 1 < nt) {   // cache-prefetch next tile (both 128B lines of this lane's 256B column)
            __builtin_prefetch(colPtr + WAVE_TILE_STRIDE, 0, 1);
            __builtin_prefetch(colPtr + WAVE_TILE_STRIDE + 64, 0, 1);
        }
        v8f c = tile_gemm(a, colPtr, koff);

        #pragma unroll
        for (int r = 0; r < 8; ++r) {
            float d  = c[r];
            float nm = fmaxf(mrun[r], d);
            srun[r]  = srun[r] * EXP2(mrun[r] - nm) + EXP2(d - nm);
            mrun[r]  = nm;
        }
        colPtr += WAVE_TILE_STRIDE;
    }

    // reduce across the 16 lanes of each half-wave (masks stay within halves)
    #pragma unroll
    for (int mask = 1; mask < 16; mask <<= 1) {
        #pragma unroll
        for (int r = 0; r < 8; ++r) {
            float om = __shfl_xor(mrun[r], mask, 32);
            float os = __shfl_xor(srun[r], mask, 32);
            float M  = fmaxf(mrun[r], om);
            float s  = 0.f;
            if (srun[r] > 0.f) s += srun[r] * EXP2(mrun[r] - M);
            if (os      > 0.f) s += os      * EXP2(om      - M);
            mrun[r] = M; srun[r] = s;
        }
    }

    if ((lane & 15) == 0) {
        int hi8 = hi ? 8 : 0;
        int p   = blockIdx.x * 8 + wave;
        #pragma unroll
        for (int r = 0; r < 8; ++r) {
            int row = m0 + r + hi8;
            partials[(size_t)row * NPART + p] = make_float2(mrun[r], srun[r]);
        }
    }
}

// ---------------- phase 3: combine partials -> per-row (M, 1/Z), base-2 domain -------
__global__ void reduce_kernel(const float2* __restrict__ partials,
                              float2* __restrict__ rowstats) {
    int row = blockIdx.x * blockDim.x + threadIdx.x;
    if (row >= BATCH) return;
    float M = -INFINITY, S = 0.f;
    const float2* p = partials + (size_t)row * NPART;
    for (int i = 0; i < NPART; ++i) {
        float2 ms = p[i];
        if (ms.y <= 0.f) continue;
        float nM = fmaxf(M, ms.x);
        S = S * EXP2(M - nM) + ms.y * EXP2(ms.x - nM);
        M = nM;
    }
    rowstats[row] = make_float2(M, 1.f / S);
}

// ---------------- phase 4: recompute logits, emit softmax (single 1.6GB store) -------
__global__ void softmax_kernel(const __bf16* __restrict__ projB,
                               const __bf16* __restrict__ outT,
                               const float2* __restrict__ rowstats,
                               float* __restrict__ out) {
    __shared__ float sM[16], sZ[16];
    int lane = threadIdx.x & 31;
    int wave = threadIdx.x >> 5;
    int m0   = blockIdx.y * 16;
    if (threadIdx.x < 16) {
        float2 ms = rowstats[m0 + threadIdx.x];
        sM[threadIdx.x] = ms.x;
        sZ[threadIdx.x] = ms.y;
    }
    __syncthreads();

    const bool hi  = lane >= 16;
    const int  hi8 = hi ? 8 : 0;
    const int koff = hi ? 16 : 0;
    const int col0 = lane & 15;

    v16bf a[4];
    load_afrag(projB, m0, lane, a);

    int base = blockIdx.x * TILES_PER_BLK + wave;
    int nt   = (base < NTILE_N) ? min(16, (NTILE_N - base + 7) / 8) : 0;

    const __bf16* colPtr = outT + (size_t)(base * 16 + col0) * EMBED;
    int ncol = base * 16 + col0;

    #pragma unroll 1
    for (int i = 0; i < nt; ++i) {
        if (i + 1 < nt) {
            __builtin_prefetch(colPtr + WAVE_TILE_STRIDE, 0, 1);
            __builtin_prefetch(colPtr + WAVE_TILE_STRIDE + 64, 0, 1);
        }
        v8f c = tile_gemm(a, colPtr, koff);

        #pragma unroll
        for (int r = 0; r < 8; ++r) {
            int row = m0 + r + hi8;
            float v = EXP2(c[r] - sM[r + hi8]) * sZ[r + hi8];
            out[(size_t)row * VOCAB + ncol] = v;
        }
        colPtr += WAVE_TILE_STRIDE;
        ncol   += 8 * 16;
    }
}

// ---------------- launch ----------------
extern "C" void kernel_launch(void* const* d_in, const int* in_sizes, int n_in,
                              void* d_out, int out_size, void* d_ws, size_t ws_size,
                              hipStream_t stream) {
    const int*   ids       = (const int*)d_in[0];     // [4096][20]
    const float* in_table  = (const float*)d_in[1];   // [100000][128]
    const float* out_table = (const float*)d_in[2];   // [128][100000]
    float*       out       = (float*)d_out;           // [4096][100000]

    // workspace layout (all offsets 256B-aligned); total ~39.6 MB
    char* ws = (char*)d_ws;
    __bf16*  projB    = (__bf16*)(ws);                                    //  1.05 MB
    __bf16*  outT     = (__bf16*)(ws + (size_t)BATCH * EMBED * 2);        // 25.60 MB
    float2*  partials = (float2*)(ws + (size_t)BATCH * EMBED * 2
                                     + (size_t)VOCAB * EMBED * 2);        // 12.85 MB
    float2*  rowstats = (float2*)(ws + (size_t)BATCH * EMBED * 2
                                     + (size_t)VOCAB * EMBED * 2
                                     + (size_t)BATCH * NPART * sizeof(float2));

    proj_kernel<<<BATCH, EMBED, 0, stream>>>(ids, in_table, projB);
    cvt_kernel<<<VOCAB / 32, 256, 0, stream>>>(out_table, outT);

    dim3 gGemm(NCHUNK, BATCH / 16);       // 49 x 256
    partial_kernel<<<gGemm, 256, 0, stream>>>(projB, outT, partials);
    reduce_kernel<<<BATCH / 256, 256, 0, stream>>>(partials, rowstats);
    softmax_kernel<<<gGemm, 256, 0, stream>>>(projB, outT, rowstats, out);
}
